// DualGraphTransformer_53944789238507
// MI455X (gfx1250) — compile-verified, matrix operand order checked by reading, not run on hardware
//
#include <hip/hip_runtime.h>
#include <hip/hip_bf16.h>

typedef __attribute__((ext_vector_type(16))) _Float16 v16h;
typedef __attribute__((ext_vector_type(8)))  _Float16 v8h;
typedef __attribute__((ext_vector_type(8)))  float    v8f;

#define DIM 128
#define TPB 256

// ---------------------------------------------------------------------------
// WMMA 16x16x32 f16->f32 (wave32). Layouts per CDNA5 ISA 7.12.2:
//  A: lane m = lane&15, half = lane>>4; element e -> k = (e<8 ? half*8+e
//     : 16+half*8+e-8).  B mirrored (lane gives N).  C/D: VGPR r ->
//     M = r + 8*(lane>>4), N = lane&15.
// ---------------------------------------------------------------------------
__device__ __forceinline__ int frag_k(int vg, int half, int j) {
  return (vg < 4 ? half * 8 + 2 * vg : 16 + half * 8 + 2 * (vg - 4)) + j;
}

__device__ __forceinline__ v8f wmma_f16(v16h a, v16h b, v8f c) {
  return __builtin_amdgcn_wmma_f32_16x16x32_f16(false, a, false, b, (short)0, c, false, false);
}

__device__ __forceinline__ v16h cvt_a(float4 f0, float4 f1, float4 f2, float4 f3) {
  v16h a;
  a[0]  = (_Float16)f0.x; a[1]  = (_Float16)f0.y; a[2]  = (_Float16)f0.z; a[3]  = (_Float16)f0.w;
  a[4]  = (_Float16)f1.x; a[5]  = (_Float16)f1.y; a[6]  = (_Float16)f1.z; a[7]  = (_Float16)f1.w;
  a[8]  = (_Float16)f2.x; a[9]  = (_Float16)f2.y; a[10] = (_Float16)f2.z; a[11] = (_Float16)f2.w;
  a[12] = (_Float16)f3.x; a[13] = (_Float16)f3.y; a[14] = (_Float16)f3.z; a[15] = (_Float16)f3.w;
  return a;
}

// A fragment from a global f32 row: 4 x 16B vector loads + packed convert.
__device__ __forceinline__ v16h load_a_g(const float* row, int half) {
  const float4* p = (const float4*)(row + half * 8);
  const float4* q = (const float4*)(row + 16 + half * 8);
  return cvt_a(p[0], p[1], q[0], q[1]);
}

__device__ __forceinline__ v16h load_a_g_scaled(const float* row, int half, float sc) {
  const float4* p = (const float4*)(row + half * 8);
  const float4* q = (const float4*)(row + 16 + half * 8);
  float4 f0 = p[0], f1 = p[1], f2 = q[0], f3 = q[1];
  f0.x *= sc; f0.y *= sc; f0.z *= sc; f0.w *= sc;
  f1.x *= sc; f1.y *= sc; f1.z *= sc; f1.w *= sc;
  f2.x *= sc; f2.y *= sc; f2.z *= sc; f2.w *= sc;
  f3.x *= sc; f3.y *= sc; f3.z *= sc; f3.w *= sc;
  return cvt_a(f0, f1, f2, f3);
}

// A fragment from an f16 row (LDS or global): two contiguous 16B chunks.
__device__ __forceinline__ v16h load_a_h(const _Float16* row, int half) {
  v8h c0 = *(const v8h*)(row + half * 8);
  v8h c1 = *(const v8h*)(row + 16 + half * 8);
  v16h a;
#pragma unroll
  for (int e = 0; e < 8; e++) { a[e] = c0[e]; a[8 + e] = c1[e]; }
  return a;
}

// B fragment from pre-swizzled weights: one contiguous 32B per-lane load,
// coalesced across the wave (1 KB per wave per K-step).
__device__ __forceinline__ v16h load_b_g(const _Float16* __restrict__ Wp, int kb, int wave,
                                         int lane) {
  return *(const v16h*)(Wp + ((((size_t)kb * 8 + wave) * 32 + lane) << 4));
}

__device__ __forceinline__ void atomicMaxF(float* addr, float val) {
  unsigned int* u = (unsigned int*)addr;
  unsigned int old = *u;
  while (__uint_as_float(old) < val) {
    unsigned int assumed = old;
    old = atomicCAS(u, assumed, __float_as_uint(val));
    if (old == assumed) break;
  }
}

// ---------------------------------------------------------------------------
// One-time weight repack: W[K x 128] f32 -> fragment-ready f16 (zero-padded).
// Wp index = ((kb*8 + cb)*32 + lane)*16 + e ; value = W[kb*32 + frag_k][cb*16 + n]
// ---------------------------------------------------------------------------
__global__ void pack_w_kernel(const float* __restrict__ W, _Float16* __restrict__ Wp,
                              int K, int Kpad) {
  int i = blockIdx.x * blockDim.x + threadIdx.x;
  if (i >= Kpad * DIM) return;
  int e = i & 15, lane = (i >> 4) & 31, cb = (i >> 9) & 7, kb = i >> 12;
  int k = kb * 32 + frag_k(e >> 1, lane >> 4, e & 1);
  int n = cb * 16 + (lane & 15);
  Wp[i] = (_Float16)((k < K) ? W[k * DIM + n] : 0.f);
}

// One-time input pads to 32-wide rows (kills all guarded scalar-load paths).
__global__ void pad_rows_f_kernel(const float* __restrict__ src, int K,
                                  float* __restrict__ dst, int n) {
  int i = blockIdx.x * blockDim.x + threadIdx.x;
  if (i >= n * 32) return;
  int r = i >> 5, c = i & 31;
  dst[i] = (c < K) ? src[(size_t)r * K + c] : 0.f;
}

__global__ void pad_rows_h_kernel(const float* __restrict__ src, int K,
                                  _Float16* __restrict__ dst, int n) {
  int i = blockIdx.x * blockDim.x + threadIdx.x;
  if (i >= n * 32) return;
  int r = i >> 5, c = i & 31;
  dst[i] = (_Float16)((c < K) ? src[(size_t)r * K + c] : 0.f);
}

// ---------------------------------------------------------------------------
// GEMM: out[M x 128] = act(A[M x K] @ W + bias) (+residual).  K % 32 == 0.
// 256 thr = 8 waves; block covers 32 rows: each wave loads B once per K-step
// and issues two WMMAs (two 16-row tiles) -> half the B traffic.  Tail rows
// are handled by clamping the row index (always-load) and guarding stores,
// keeping the K-loop free of exec-mask juggling.
// act: 0 none, 1 SiLU, 2 ReLU
// ---------------------------------------------------------------------------
__global__ void gemm_n128_kernel(const float* __restrict__ A, const _Float16* __restrict__ Wp,
                                 const float* __restrict__ bias, const float* __restrict__ residual,
                                 float* __restrict__ out, int M, int K, int act) {
  int tid = threadIdx.x, lane = tid & 31, wave = tid >> 5, half = lane >> 4;
  int row0 = blockIdx.x * 32, co = wave * 16;
  int m0 = row0 + (lane & 15), m1 = m0 + 16;
  int mc0 = (m0 < M) ? m0 : (M - 1);
  int mc1 = (m1 < M) ? m1 : (M - 1);
  const float* ar0 = A + (size_t)mc0 * K;
  const float* ar1 = A + (size_t)mc1 * K;
  v8f acc0 = {}, acc1 = {};
  int nkb = K >> 5;
  for (int kb = 0; kb < nkb; kb++) {
    int k0 = kb << 5;
    v16h b = load_b_g(Wp, kb, wave, lane);
    v16h a0 = load_a_g(ar0 + k0, half);
    v16h a1 = load_a_g(ar1 + k0, half);
    acc0 = wmma_f16(a0, b, acc0);
    acc1 = wmma_f16(a1, b, acc1);
  }
  int n = co + (lane & 15), mb = half * 8;
  float bi = bias[n];
#pragma unroll
  for (int r = 0; r < 8; r++) {
    int mm = row0 + mb + r;
    if (mm < M) {
      float v = acc0[r] + bi;
      if (act == 1) v = v / (1.f + __expf(-v));
      else if (act == 2) v = fmaxf(v, 0.f);
      if (residual) v += residual[(size_t)mm * DIM + n];
      out[(size_t)mm * DIM + n] = v;
    }
    int mm1 = mm + 16;
    if (mm1 < M) {
      float v = acc1[r] + bi;
      if (act == 1) v = v / (1.f + __expf(-v));
      else if (act == 2) v = fmaxf(v, 0.f);
      if (residual) v += residual[(size_t)mm1 * DIM + n];
      out[(size_t)mm1 * DIM + n] = v;
    }
  }
}

// ---------------------------------------------------------------------------
// Fused EGNN edge kernel (16 edges/block), K-blocks segment-aligned:
//   kb 0-3: x[src], kb 4-7: x[dst], kb 8: padded f16 edge-attr (stride 32).
//   hidden = SiLU(. @ W1 + b1);  e = hidden @ W2 + b2;
//   coeff = tanh(e . cw + cb); scatter e/cnt/coeff*dpos via f32 atomics.
// ---------------------------------------------------------------------------
__global__ void egnn_edge_kernel(const float* __restrict__ x, const float* __restrict__ pos,
                                 const int* __restrict__ src, const int* __restrict__ dst,
                                 const _Float16* __restrict__ eap, int E,
                                 const _Float16* __restrict__ W1p, const float* __restrict__ b1,
                                 const _Float16* __restrict__ W2p, const float* __restrict__ b2,
                                 const float* __restrict__ cw, const float* __restrict__ cbp,
                                 float* __restrict__ msum, float* __restrict__ cnt,
                                 float* __restrict__ possum) {
  __shared__ _Float16 sH[16 * DIM];
  __shared__ float sE[16 * DIM];
  __shared__ int sS[16], sD[16];
  int tid = threadIdx.x, lane = tid & 31, wave = tid >> 5, half = lane >> 4;
  int e0 = blockIdx.x * 16, co = wave * 16;
  if (tid < 16) {
    int e = e0 + tid;
    sS[tid] = (e < E) ? src[e] : 0;
    sD[tid] = (e < E) ? dst[e] : 0;
  }
  __syncthreads();
  int mrow = lane & 15;
  int s = sS[mrow], d = sD[mrow];
  int eidx = e0 + mrow;
  int eic = (eidx < E) ? eidx : (E - 1);
  const float* xs = x + (size_t)s * DIM;
  const float* xd = x + (size_t)d * DIM;
  const _Float16* er = eap + (size_t)eic * 32;
  v8f acc = {};
#pragma unroll
  for (int kb = 0; kb < 9; kb++) {
    int k0 = kb << 5;
    v16h a;
    if (k0 < DIM) a = load_a_g(xs + k0, half);
    else if (k0 < 2 * DIM) a = load_a_g(xd + (k0 - DIM), half);
    else a = load_a_h(er, half);
    acc = wmma_f16(a, load_b_g(W1p, kb, wave, lane), acc);
  }
  {
    int n = co + mrow, mb = half * 8;
#pragma unroll
    for (int r = 0; r < 8; r++) {
      float v = acc[r] + b1[n];
      v = v / (1.f + __expf(-v));  // SiLU
      sH[(mb + r) * DIM + n] = (_Float16)v;
    }
  }
  __syncthreads();
  const _Float16* hrow = sH + mrow * DIM;
  v8f acc2 = {};
#pragma unroll
  for (int kb = 0; kb < 4; kb++) {
    acc2 = wmma_f16(load_a_h(hrow + kb * 32, half), load_b_g(W2p, kb, wave, lane), acc2);
  }
  {
    int n = co + mrow, mb = half * 8;
#pragma unroll
    for (int r = 0; r < 8; r++) sE[(mb + r) * DIM + n] = acc2[r] + b2[n];
  }
  __syncthreads();
  if (tid < 16) {
    int e = e0 + tid;
    if (e < E) {
      float sum = cbp[0];
      for (int c = 0; c < DIM; c++) sum += sE[tid * DIM + c] * cw[c];
      float coef = tanhf(sum);
      int si = sS[tid], di = sD[tid];
      atomicAdd(&cnt[di], 1.f);
#pragma unroll
      for (int c = 0; c < 3; c++)
        atomicAdd(&possum[di * 3 + c], coef * (pos[si * 3 + c] - pos[di * 3 + c]));
    }
  }
  for (int i = tid; i < 16 * DIM; i += TPB) {
    int r = i >> 7, e = e0 + r;
    if (e < E) atomicAdd(&msum[(size_t)sD[r] * DIM + (i & 127)], sE[i]);
  }
}

// ---------------------------------------------------------------------------
// Fused EGNN node kernel: out = SiLU([x | msum/cnt] @ W1 + b1) @ W2 + b2
// ---------------------------------------------------------------------------
__global__ void egnn_node_kernel(const float* __restrict__ x, const float* __restrict__ msum,
                                 const float* __restrict__ cnt,
                                 const _Float16* __restrict__ W1p, const float* __restrict__ b1,
                                 const _Float16* __restrict__ W2p, const float* __restrict__ b2,
                                 float* __restrict__ out, int M) {
  __shared__ _Float16 sH[16 * DIM];
  int tid = threadIdx.x, lane = tid & 31, wave = tid >> 5, half = lane >> 4;
  int row0 = blockIdx.x * 16, co = wave * 16;
  int mrow = lane & 15;
  int m = row0 + mrow;
  int mc = (m < M) ? m : (M - 1);
  const float* xr = x + (size_t)mc * DIM;
  const float* mr = msum + (size_t)mc * DIM;
  float rc = 1.f / fmaxf(cnt[mc], 1.f);
  v8f acc = {};
#pragma unroll
  for (int kb = 0; kb < 8; kb++) {
    int k0 = kb << 5;
    v16h a = (k0 < DIM) ? load_a_g(xr + k0, half) : load_a_g_scaled(mr + (k0 - DIM), half, rc);
    acc = wmma_f16(a, load_b_g(W1p, kb, wave, lane), acc);
  }
  {
    int n = co + mrow, mb = half * 8;
#pragma unroll
    for (int r = 0; r < 8; r++) {
      float v = acc[r] + b1[n];
      v = v / (1.f + __expf(-v));
      sH[(mb + r) * DIM + n] = (_Float16)v;
    }
  }
  __syncthreads();
  const _Float16* hrow = sH + mrow * DIM;
  v8f acc2 = {};
#pragma unroll
  for (int kb = 0; kb < 4; kb++) {
    acc2 = wmma_f16(load_a_h(hrow + kb * 32, half), load_b_g(W2p, kb, wave, lane), acc2);
  }
  int n = co + mrow, mb = half * 8;
#pragma unroll
  for (int r = 0; r < 8; r++) {
    int mm = row0 + mb + r;
    if (mm < M) out[(size_t)mm * DIM + n] = acc2[r] + b2[n];
  }
}

// ---------------------------------------------------------------------------
// Small utility kernels
// ---------------------------------------------------------------------------
__global__ void fill_kernel(float* p, float v, int n) {
  int i = blockIdx.x * blockDim.x + threadIdx.x;
  if (i < n) p[i] = v;
}

__global__ void pos_update_kernel(const float* pos, const float* psum, const float* cnt,
                                  float* out, int n) {
  int i = blockIdx.x * blockDim.x + threadIdx.x;
  if (i >= n * 3) return;
  int r = i / 3;
  out[i] = pos[i] + psum[i] / fmaxf(cnt[r], 1.f);
}

__global__ void split_edges_kernel(const int* ce, int* li, int* pi, int NC, int NL) {
  int i = blockIdx.x * blockDim.x + threadIdx.x;
  if (i >= NC) return;
  li[i] = ce[2 * i];
  pi[i] = ce[2 * i + 1] - NL;
}

__global__ void rbf_kernel(const float* __restrict__ a, const float* __restrict__ w,
                           const float* __restrict__ b, float* __restrict__ out, int E) {
  int i = blockIdx.x * blockDim.x + threadIdx.x;
  if (i >= E * 8) return;
  int e = i >> 3, h = i & 7;
  const float4* ap = (const float4*)(a + (size_t)e * 16);
  float s = b[h];
#pragma unroll
  for (int d = 0; d < 4; d++) {
    float4 f = ap[d];
    s += f.x * w[(d * 4 + 0) * 8 + h] + f.y * w[(d * 4 + 1) * 8 + h] +
         f.z * w[(d * 4 + 2) * 8 + h] + f.w * w[(d * 4 + 3) * 8 + h];
  }
  out[i] = s;
}

__global__ void attn_logits_kernel(const float* __restrict__ q, const float* __restrict__ kk,
                                   const int* __restrict__ qi, const int* __restrict__ ki,
                                   const float* __restrict__ rbf, float* __restrict__ logits,
                                   float* __restrict__ mx, int E) {
  int i = blockIdx.x * blockDim.x + threadIdx.x;
  if (i >= E * 8) return;
  int e = i >> 3, h = i & 7;
  const float4* qp = (const float4*)(q + (size_t)qi[e] * DIM + h * 16);
  const float4* kp = (const float4*)(kk + (size_t)ki[e] * DIM + h * 16);
  float s = 0.f;
#pragma unroll
  for (int d = 0; d < 4; d++) {
    float4 a = qp[d], b = kp[d];
    s += a.x * b.x + a.y * b.y + a.z * b.z + a.w * b.w;
  }
  s = s * 0.25f + rbf[i];  // SCALE = 16^-0.5
  logits[i] = s;
  atomicMaxF(&mx[qi[e] * 8 + h], s);
}

__global__ void attn_expsum_kernel(const int* __restrict__ qi, const float* __restrict__ mx,
                                   float* __restrict__ logits, float* __restrict__ sum, int E) {
  int i = blockIdx.x * blockDim.x + threadIdx.x;
  if (i >= E * 8) return;
  int e = i >> 3, h = i & 7;
  float ex = __expf(logits[i] - mx[qi[e] * 8 + h]);
  logits[i] = ex;
  atomicAdd(&sum[qi[e] * 8 + h], ex);
}

__global__ void attn_agg_kernel(const int* __restrict__ qi, const int* __restrict__ ki,
                                const float* __restrict__ logits, const float* __restrict__ sum,
                                const float* __restrict__ v, float* __restrict__ agg, int E) {
  int i = blockIdx.x * blockDim.x + threadIdx.x;
  if (i >= E * 8) return;
  int e = i >> 3, h = i & 7;
  float w = logits[i] / (sum[qi[e] * 8 + h] + 1e-8f);
  const float4* vp = (const float4*)(v + (size_t)ki[e] * DIM + h * 16);
  float* ap = agg + (size_t)qi[e] * DIM + h * 16;
#pragma unroll
  for (int j = 0; j < 4; j++) {
    float4 f = vp[j];
    atomicAdd(&ap[j * 4 + 0], w * f.x);
    atomicAdd(&ap[j * 4 + 1], w * f.y);
    atomicAdd(&ap[j * 4 + 2], w * f.z);
    atomicAdd(&ap[j * 4 + 3], w * f.w);
  }
}

__global__ void seg_pool_kernel(const float* __restrict__ x, const int* __restrict__ batch,
                                float* __restrict__ ps, float* __restrict__ pc, int n) {
  int i = blockIdx.x * blockDim.x + threadIdx.x;
  if (i >= n * DIM) return;
  int r = i >> 7, c = i & 127, b = batch[r];
  atomicAdd(&ps[b * DIM + c], x[i]);
  if (c == 0) atomicAdd(&pc[b], 1.f);
}

// Final head: one block per batch element (B = 64), 128 threads.
__global__ void head_kernel(const float* lsum, const float* lcnt, const float* psum,
                            const float* pcnt,
                            const float* p1w, const float* p1b, const float* p2w, const float* p2b,
                            const float* r1w, const float* r1b, const float* r2w, const float* r2b,
                            const float* v1w, const float* v1b, const float* v2w, const float* v2b,
                            float* out) {
  __shared__ float z[256], t1[128], t2[64], t3[32];
  int b = blockIdx.x, tid = threadIdx.x;
  if (tid < 128) {
    z[tid] = lsum[b * 128 + tid] / fmaxf(lcnt[b], 1.f);
    z[128 + tid] = psum[b * 128 + tid] / fmaxf(pcnt[b], 1.f);
  }
  __syncthreads();
  if (tid < 128) {
    float s = p1b[tid];
    for (int k = 0; k < 256; k++) s += z[k] * p1w[k * 128 + tid];
    t1[tid] = fmaxf(s, 0.f);
  }
  __syncthreads();
  if (tid < 64) {
    float s = p2b[tid];
    for (int k = 0; k < 128; k++) s += t1[k] * p2w[k * 64 + tid];
    t2[tid] = fmaxf(s, 0.f);
  }
  __syncthreads();
  if (tid < 32) {
    float s = r1b[tid];
    for (int k = 0; k < 64; k++) s += t2[k] * r1w[k * 32 + tid];
    t3[tid] = fmaxf(s, 0.f);
  }
  __syncthreads();
  if (tid == 0) {
    float s = r2b[0];
    for (int k = 0; k < 32; k++) s += t3[k] * r2w[k];
    out[b] = s;  // mu
  }
  __syncthreads();
  if (tid < 32) {
    float s = v1b[tid];
    for (int k = 0; k < 64; k++) s += t2[k] * v1w[k * 32 + tid];
    t3[tid] = fmaxf(s, 0.f);
  }
  __syncthreads();
  if (tid == 0) {
    float s = v2b[0];
    for (int k = 0; k < 32; k++) s += t3[k] * v2w[k];
    out[64 + b] = s;  // logvar
  }
}

// ---------------------------------------------------------------------------
// Orchestration
// ---------------------------------------------------------------------------
extern "C" void kernel_launch(void* const* d_in, const int* in_sizes, int n_in,
                              void* d_out, int out_size, void* d_ws, size_t ws_size,
                              hipStream_t stream) {
  (void)in_sizes; (void)n_in; (void)out_size; (void)ws_size;
  const int NL = 20000, NP = 50000, ELIG = 160000, EPROT = 400000, NC = 250000;

  const float* lig_x    = (const float*)d_in[0];
  const float* lig_pos  = (const float*)d_in[1];
  const int*   lig_ei   = (const int*)d_in[2];
  const float* lig_ea   = (const float*)d_in[3];
  const int*   lig_b    = (const int*)d_in[4];
  const float* prot_x   = (const float*)d_in[5];
  const float* prot_pos = (const float*)d_in[6];
  const int*   prot_ei  = (const int*)d_in[7];
  const float* prot_ea  = (const float*)d_in[8];
  const int*   prot_b   = (const int*)d_in[9];
  const int*   ce       = (const int*)d_in[10];
  const float* cattr    = (const float*)d_in[11];
  // Param leaves (JAX pytree: sorted dict keys, 'b' before 'w'):
  // 0-1 attn_out, 2-41 lig_egnn[0..3]{c,e1,e2,n1,n2}, 42-43 lig_proj,
  // 44-47 lig_read1/2, 48-51 lv1/2, 52-55 pool1/2, 56-59 pro_read1/2,
  // 60-99 prot_egnn, 100-101 prot_proj, 102-105 reg1/2, 106-107 rbf_proj,
  // 108-113 to_k/to_q/to_v.
  void* const* P = d_in + 12;
  auto PF = [&](int i) -> const float* { return (const float*)P[i]; };

  char* base = (char*)d_ws;
  size_t off = 0;
  auto alloc = [&](size_t elems) -> float* {
    float* p = (float*)(base + off);
    off = (off + elems * 4 + 255) & ~(size_t)255;
    return p;
  };
  auto alloci = [&](size_t elems) -> int* {
    int* p = (int*)(base + off);
    off = (off + elems * 4 + 255) & ~(size_t)255;
    return p;
  };
  auto alloch = [&](size_t elems) -> _Float16* {
    _Float16* p = (_Float16*)(base + off);
    off = (off + elems * 2 + 255) & ~(size_t)255;
    return p;
  };

  dim3 B256(TPB);
  auto g16 = [](int m) { return dim3((m + 15) / 16); };
  auto g32 = [](int m) { return dim3((m + 31) / 32); };
  auto gn  = [](int n) { return dim3((n + TPB - 1) / TPB); };

  // One-time weight repack to fragment-ready f16
  auto pack = [&](const float* W, int K) -> const _Float16* {
    int Kpad = (K + 31) / 32 * 32;
    _Float16* Wp = alloch((size_t)Kpad * DIM);
    pack_w_kernel<<<gn(Kpad * DIM), B256, 0, stream>>>(W, Wp, K, Kpad);
    return Wp;
  };

  const _Float16* Wq  = pack(PF(111), DIM);
  const _Float16* Wk  = pack(PF(109), DIM);
  const _Float16* Wv  = pack(PF(113), DIM);
  const _Float16* Wo  = pack(PF(1), DIM);
  const _Float16* Wlp = pack(PF(43), 11);
  const _Float16* Wpp = pack(PF(101), 20);
  const _Float16* Wlr1 = pack(PF(45), DIM);
  const _Float16* Wlr2 = pack(PF(47), DIM);
  const _Float16* Wpr1 = pack(PF(57), DIM);
  const _Float16* Wpr2 = pack(PF(59), DIM);
  const _Float16 *We1l[4], *We2l[4], *Wn1l[4], *Wn2l[4];
  const _Float16 *We1p[4], *We2p[4], *Wn1p[4], *Wn2p[4];
  for (int t = 0; t < 4; t++) {
    int lb = 2 + 10 * t, pb = 60 + 10 * t;
    We1l[t] = pack(PF(lb + 3), 2 * DIM + 6);
    We2l[t] = pack(PF(lb + 5), DIM);
    Wn1l[t] = pack(PF(lb + 7), 2 * DIM);
    Wn2l[t] = pack(PF(lb + 9), DIM);
    We1p[t] = pack(PF(pb + 3), 2 * DIM + 16);
    We2p[t] = pack(PF(pb + 5), DIM);
    Wn1p[t] = pack(PF(pb + 7), 2 * DIM);
    Wn2p[t] = pack(PF(pb + 9), DIM);
  }

  // One-time input pads
  float* xlp = alloc((size_t)NL * 32);
  float* xpp = alloc((size_t)NP * 32);
  _Float16* elp = alloch((size_t)ELIG * 32);
  _Float16* epp = alloch((size_t)EPROT * 32);
  pad_rows_f_kernel<<<gn(NL * 32), B256, 0, stream>>>(lig_x, 11, xlp, NL);
  pad_rows_f_kernel<<<gn(NP * 32), B256, 0, stream>>>(prot_x, 20, xpp, NP);
  pad_rows_h_kernel<<<gn(ELIG * 32), B256, 0, stream>>>(lig_ea, 6, elp, ELIG);
  pad_rows_h_kernel<<<gn(EPROT * 32), B256, 0, stream>>>(prot_ea, 16, epp, EPROT);

  float* hA_l = alloc((size_t)NL * DIM); float* hB_l = alloc((size_t)NL * DIM);
  float* hA_p = alloc((size_t)NP * DIM); float* hB_p = alloc((size_t)NP * DIM);
  float* pA_l = alloc(NL * 3); float* pB_l = alloc(NL * 3);
  float* pA_p = alloc(NP * 3); float* pB_p = alloc(NP * 3);
  float* msum_l = alloc((size_t)NL * DIM); float* msum_p = alloc((size_t)NP * DIM);
  float* cnt_l = alloc(NL); float* cnt_p = alloc(NP);
  float* psm_l = alloc(NL * 3); float* psm_p = alloc(NP * 3);
  float* q_l = alloc((size_t)NL * DIM); float* k_l = alloc((size_t)NL * DIM);
  float* v_l = alloc((size_t)NL * DIM);
  float* q_p = alloc((size_t)NP * DIM); float* k_p = alloc((size_t)NP * DIM);
  float* v_p = alloc((size_t)NP * DIM);
  float* rbf = alloc((size_t)NC * 8); float* logits = alloc((size_t)NC * 8);
  float* mx_l = alloc(NL * 8); float* sm_l = alloc(NL * 8);
  float* mx_p = alloc(NP * 8); float* sm_p = alloc(NP * 8);
  int* li = alloci(NC); int* pi = alloci(NC);
  float* pls = alloc(64 * DIM); float* plc = alloc(64);
  float* pps = alloc(64 * DIM); float* ppc = alloc(64);
  float* agg_l = msum_l;  // alias: msum dead by attention phase
  float* agg_p = msum_p;

  // Initial projections (padded inputs, K = 32)
  gemm_n128_kernel<<<g32(NL), B256, 0, stream>>>(xlp, Wlp, PF(42), nullptr, hA_l, NL, 32, 0);
  gemm_n128_kernel<<<g32(NP), B256, 0, stream>>>(xpp, Wpp, PF(100), nullptr, hA_p, NP, 32, 0);
  split_edges_kernel<<<gn(NC), B256, 0, stream>>>(ce, li, pi, NC, NL);
  rbf_kernel<<<gn(NC * 8), B256, 0, stream>>>(cattr, PF(107), PF(106), rbf, NC);

  const float* pcur_l = lig_pos; const float* pcur_p = prot_pos;
  float* hcur_l = hA_l; float* hnext_l = hB_l;
  float* hcur_p = hA_p; float* hnext_p = hB_p;
  float* pbuf_l[2] = {pA_l, pB_l}; float* pbuf_p[2] = {pA_p, pB_p};

  for (int t = 0; t < 4; t++) {
    int lb = 2 + 10 * t, pb = 60 + 10 * t;
    fill_kernel<<<gn(NL * DIM), B256, 0, stream>>>(msum_l, 0.f, NL * DIM);
    fill_kernel<<<gn(NP * DIM), B256, 0, stream>>>(msum_p, 0.f, NP * DIM);
    fill_kernel<<<gn(NL), B256, 0, stream>>>(cnt_l, 0.f, NL);
    fill_kernel<<<gn(NP), B256, 0, stream>>>(cnt_p, 0.f, NP);
    fill_kernel<<<gn(NL * 3), B256, 0, stream>>>(psm_l, 0.f, NL * 3);
    fill_kernel<<<gn(NP * 3), B256, 0, stream>>>(psm_p, 0.f, NP * 3);

    egnn_edge_kernel<<<g16(ELIG), B256, 0, stream>>>(
        hcur_l, pcur_l, lig_ei, lig_ei + ELIG, elp, ELIG,
        We1l[t], PF(lb + 2), We2l[t], PF(lb + 4), PF(lb + 1), PF(lb + 0),
        msum_l, cnt_l, psm_l);
    egnn_edge_kernel<<<g16(EPROT), B256, 0, stream>>>(
        hcur_p, pcur_p, prot_ei, prot_ei + EPROT, epp, EPROT,
        We1p[t], PF(pb + 2), We2p[t], PF(pb + 4), PF(pb + 1), PF(pb + 0),
        msum_p, cnt_p, psm_p);
    egnn_node_kernel<<<g16(NL), B256, 0, stream>>>(
        hcur_l, msum_l, cnt_l, Wn1l[t], PF(lb + 6), Wn2l[t], PF(lb + 8), hnext_l, NL);
    egnn_node_kernel<<<g16(NP), B256, 0, stream>>>(
        hcur_p, msum_p, cnt_p, Wn1p[t], PF(pb + 6), Wn2p[t], PF(pb + 8), hnext_p, NP);
    pos_update_kernel<<<gn(NL * 3), B256, 0, stream>>>(pcur_l, psm_l, cnt_l, pbuf_l[t & 1], NL);
    pos_update_kernel<<<gn(NP * 3), B256, 0, stream>>>(pcur_p, psm_p, cnt_p, pbuf_p[t & 1], NP);
    pcur_l = pbuf_l[t & 1]; pcur_p = pbuf_p[t & 1];

    // QKV projections
    gemm_n128_kernel<<<g32(NL), B256, 0, stream>>>(hnext_l, Wq, PF(110), nullptr, q_l, NL, DIM, 0);
    gemm_n128_kernel<<<g32(NL), B256, 0, stream>>>(hnext_l, Wk, PF(108), nullptr, k_l, NL, DIM, 0);
    gemm_n128_kernel<<<g32(NL), B256, 0, stream>>>(hnext_l, Wv, PF(112), nullptr, v_l, NL, DIM, 0);
    gemm_n128_kernel<<<g32(NP), B256, 0, stream>>>(hnext_p, Wq, PF(110), nullptr, q_p, NP, DIM, 0);
    gemm_n128_kernel<<<g32(NP), B256, 0, stream>>>(hnext_p, Wk, PF(108), nullptr, k_p, NP, DIM, 0);
    gemm_n128_kernel<<<g32(NP), B256, 0, stream>>>(hnext_p, Wv, PF(112), nullptr, v_p, NP, DIM, 0);

    // Ligand-side attention
    fill_kernel<<<gn(NL * 8), B256, 0, stream>>>(mx_l, -1e30f, NL * 8);
    fill_kernel<<<gn(NL * 8), B256, 0, stream>>>(sm_l, 0.f, NL * 8);
    fill_kernel<<<gn(NL * DIM), B256, 0, stream>>>(agg_l, 0.f, NL * DIM);
    attn_logits_kernel<<<gn(NC * 8), B256, 0, stream>>>(q_l, k_p, li, pi, rbf, logits, mx_l, NC);
    attn_expsum_kernel<<<gn(NC * 8), B256, 0, stream>>>(li, mx_l, logits, sm_l, NC);
    attn_agg_kernel<<<gn(NC * 8), B256, 0, stream>>>(li, pi, logits, sm_l, v_p, agg_l, NC);
    gemm_n128_kernel<<<g32(NL), B256, 0, stream>>>(agg_l, Wo, PF(0), hnext_l, hnext_l, NL, DIM, 0);

    // Protein-side attention
    fill_kernel<<<gn(NP * 8), B256, 0, stream>>>(mx_p, -1e30f, NP * 8);
    fill_kernel<<<gn(NP * 8), B256, 0, stream>>>(sm_p, 0.f, NP * 8);
    fill_kernel<<<gn(NP * DIM), B256, 0, stream>>>(agg_p, 0.f, NP * DIM);
    attn_logits_kernel<<<gn(NC * 8), B256, 0, stream>>>(q_p, k_l, pi, li, rbf, logits, mx_p, NC);
    attn_expsum_kernel<<<gn(NC * 8), B256, 0, stream>>>(pi, mx_p, logits, sm_p, NC);
    attn_agg_kernel<<<gn(NC * 8), B256, 0, stream>>>(pi, li, logits, sm_p, v_l, agg_p, NC);
    gemm_n128_kernel<<<g32(NP), B256, 0, stream>>>(agg_p, Wo, PF(0), hnext_p, hnext_p, NP, DIM, 0);

    float* tl = hcur_l; hcur_l = hnext_l; hnext_l = tl;
    float* tp = hcur_p; hcur_p = hnext_p; hnext_p = tp;
  }

  // Readouts (reuse q/k buffers as scratch — attention done)
  float* tmp_l = q_l; float* ligr = k_l; float* tmp_p = q_p; float* protr = k_p;
  gemm_n128_kernel<<<g32(NL), B256, 0, stream>>>(hcur_l, Wlr1, PF(44), nullptr, tmp_l, NL, DIM, 2);
  gemm_n128_kernel<<<g32(NL), B256, 0, stream>>>(tmp_l, Wlr2, PF(46), nullptr, ligr, NL, DIM, 0);
  gemm_n128_kernel<<<g32(NP), B256, 0, stream>>>(hcur_p, Wpr1, PF(56), nullptr, tmp_p, NP, DIM, 2);
  gemm_n128_kernel<<<g32(NP), B256, 0, stream>>>(tmp_p, Wpr2, PF(58), nullptr, protr, NP, DIM, 0);

  fill_kernel<<<gn(64 * DIM), B256, 0, stream>>>(pls, 0.f, 64 * DIM);
  fill_kernel<<<gn(64), B256, 0, stream>>>(plc, 0.f, 64);
  fill_kernel<<<gn(64 * DIM), B256, 0, stream>>>(pps, 0.f, 64 * DIM);
  fill_kernel<<<gn(64), B256, 0, stream>>>(ppc, 0.f, 64);
  seg_pool_kernel<<<gn(NL * DIM), B256, 0, stream>>>(ligr, lig_b, pls, plc, NL);
  seg_pool_kernel<<<gn(NP * DIM), B256, 0, stream>>>(protr, prot_b, pps, ppc, NP);

  head_kernel<<<dim3(64), dim3(128), 0, stream>>>(
      pls, plc, pps, ppc,
      PF(53), PF(52), PF(55), PF(54),      // pool1, pool2
      PF(103), PF(102), PF(105), PF(104),  // reg1, reg2
      PF(49), PF(48), PF(51), PF(50),      // lv1, lv2
      (float*)d_out);
}
